// DelayedDMD_206158430257
// MI455X (gfx1250) — matrix operand order repaired and checked by reading, not compiled, not run on hardware
//
#include <hip/hip_runtime.h>
#include <hip/hip_bf16.h>

// Problem dims (fixed by the reference)
constexpr int BSn = 64;
constexpr int Dd  = 256;
constexpr int Mm  = 512;
constexpr int MPROX = 10;

typedef __bf16 bf16_t;
typedef __attribute__((ext_vector_type(16))) __bf16 v16bf;
typedef __attribute__((ext_vector_type(8)))  float  v8f;
typedef __attribute__((ext_vector_type(4)))  unsigned int v4u;
typedef __attribute__((ext_vector_type(8)))  int v8i;
typedef __attribute__((ext_vector_type(4)))  int v4i;

// ---------------------------------------------------------------------------
// WMMA helper: D = A(16x32 bf16) * B(32x16 bf16) + C(16x16 f32)
// ---------------------------------------------------------------------------
__device__ inline v8f wmma_bf16(v16bf a, v16bf b, v8f c) {
  return __builtin_amdgcn_wmma_f32_16x16x32_bf16(
      /*neg_a=*/false, a, /*neg_b=*/false, b,
      /*c_mod=*/(short)0, c, /*reuse_a=*/false, /*reuse_b=*/false);
}

// ---------------------------------------------------------------------------
// Fragment loaders (wave32; A and B both use contiguous k = 16*(lane/16) + element)
// ---------------------------------------------------------------------------

// A fragment (f32 source, cvt): 16x32 (MxK), A[m][k] = src[(row0+m)*ld + k0+k]
__device__ inline v16bf load_frag_A_f32(const float* __restrict__ src, int ld,
                                        int row0, int k0) {
  const int lane = threadIdx.x & 31;
  const int row  = row0 + (lane & 15);
  const int h    = (lane >> 4) & 1;
  const float* p = src + (size_t)row * ld;
  v16bf f;
#pragma unroll
  for (int v = 0; v < 8; ++v) {
    const int k = k0 + (h << 4) + (v << 1);
    f[2 * v]     = (__bf16)p[k];
    f[2 * v + 1] = (__bf16)p[k + 1];
  }
  return f;
}

// A fragment (bf16 source, no cvt)
__device__ inline v16bf load_frag_A_bf(const bf16_t* __restrict__ src, int ld,
                                       int row0, int k0) {
  const int lane = threadIdx.x & 31;
  const int row  = row0 + (lane & 15);
  const int h    = (lane >> 4) & 1;
  const bf16_t* p = src + (size_t)row * ld;
  v16bf f;
#pragma unroll
  for (int v = 0; v < 8; ++v) {
    const int k = k0 + (h << 4) + (v << 1);
    f[2 * v]     = p[k];
    f[2 * v + 1] = p[k + 1];
  }
  return f;
}

// B fragment (row-major f32 source): 32x16 (KxN), B[k][n] = src[(k0+k)*ld + n0+n]
__device__ inline v16bf load_frag_B_rm_f32(const float* __restrict__ src, int ld,
                                           int k0, int n0) {
  const int lane = threadIdx.x & 31;
  const int col  = n0 + (lane & 15);
  const int kb   = k0 + ((lane >> 4) << 4);
  v16bf f;
#pragma unroll
  for (int v = 0; v < 8; ++v) {
    const int k = kb + 2 * v;
    f[2 * v]     = (__bf16)src[(size_t)k * ld + col];
    f[2 * v + 1] = (__bf16)src[(size_t)(k + 1) * ld + col];
  }
  return f;
}

// B fragment (transposed f32 source): B[k][n] = src[(n0+n)*ld + k0+k]
__device__ inline v16bf load_frag_B_tr_f32(const float* __restrict__ src, int ld,
                                           int k0, int n0) {
  const int lane = threadIdx.x & 31;
  const int col  = n0 + (lane & 15);
  const int kb   = k0 + ((lane >> 4) << 4);
  const float* p = src + (size_t)col * ld;
  v16bf f;
#pragma unroll
  for (int v = 0; v < 8; ++v) {
    const int k = kb + 2 * v;
    f[2 * v]     = (__bf16)p[k];
    f[2 * v + 1] = (__bf16)p[k + 1];
  }
  return f;
}

// B fragment (transposed bf16 source, contiguous K pairs, no cvt).
// Works for global or LDS-resident panels (col-major, leading dim ld).
__device__ inline v16bf load_frag_B_tr_bf(const bf16_t* __restrict__ src, int ld,
                                          int k0, int n0) {
  const int lane = threadIdx.x & 31;
  const int col  = n0 + (lane & 15);
  const int kb   = k0 + ((lane >> 4) << 4);
  const bf16_t* p = src + (size_t)col * ld;
  v16bf f;
#pragma unroll
  for (int v = 0; v < 8; ++v) {
    const int k = kb + 2 * v;
    f[2 * v]     = p[k];
    f[2 * v + 1] = p[k + 1];
  }
  return f;
}

// C/D tile (f32): lane col = lane&15, rows = row0 + (lane>>4)*8 + v
__device__ inline v8f load_tile_C(const float* __restrict__ src, int ld,
                                  int row0, int col0) {
  const int lane = threadIdx.x & 31;
  const int col  = col0 + (lane & 15);
  const int rb   = row0 + ((lane >> 4) << 3);
  v8f c;
#pragma unroll
  for (int v = 0; v < 8; ++v) c[v] = src[(size_t)(rb + v) * ld + col];
  return c;
}

__device__ inline void store_tile_C(float* __restrict__ dst, int ld,
                                    int row0, int col0, v8f acc) {
  const int lane = threadIdx.x & 31;
  const int col  = col0 + (lane & 15);
  const int rb   = row0 + ((lane >> 4) << 3);
#pragma unroll
  for (int v = 0; v < 8; ++v) dst[(size_t)(rb + v) * ld + col] = acc[v];
}

// Transposed C-store: dst[col][row] = acc  (each lane writes 8 contiguous elems)
__device__ inline void store_tile_tr_bf(bf16_t* __restrict__ dst, int ld,
                                        int row0, int col0, v8f acc) {
  const int lane = threadIdx.x & 31;
  const int col  = col0 + (lane & 15);
  const int rb   = row0 + ((lane >> 4) << 3);
  bf16_t* p = dst + (size_t)col * ld + rb;
#pragma unroll
  for (int v = 0; v < 8; ++v) p[v] = (__bf16)acc[v];
}

__device__ inline void store_tile_tr_f32(float* __restrict__ dst, int ld,
                                         int row0, int col0, v8f acc) {
  const int lane = threadIdx.x & 31;
  const int col  = col0 + (lane & 15);
  const int rb   = row0 + ((lane >> 4) << 3);
  float* p = dst + (size_t)col * ld + rb;
#pragma unroll
  for (int v = 0; v < 8; ++v) p[v] = acc[v];
}

// ---------------------------------------------------------------------------
// Stage 1: Q[b] = P[b] (DxD) @ Y1[b] (DxM); wave computes 16x64 (4 accs)
// ---------------------------------------------------------------------------
__global__ void k_gemm_Q(const float* __restrict__ P,
                         const float* __restrict__ Y1,
                         float* __restrict__ Q) {
  const int TI = Dd / 16, TJ4 = Mm / 64;
  const int gw = (blockIdx.x * blockDim.x + threadIdx.x) >> 5;
  const int b  = gw / (TI * TJ4);
  const int r  = gw % (TI * TJ4);
  const int ti = r / TJ4, tj = (r % TJ4) * 4;   // tile-col base, 16-col units
  const float* Pb = P  + (size_t)b * Dd * Dd;
  const float* Yb = Y1 + (size_t)b * Dd * Mm;
  float*       Qb = Q  + (size_t)b * Dd * Mm;

  v8f acc[4] = {{}, {}, {}, {}};
#pragma unroll 4
  for (int k = 0; k < Dd; k += 32) {
    v16bf a = load_frag_A_f32(Pb, Dd, ti * 16, k);
#pragma unroll
    for (int t = 0; t < 4; ++t) {
      v16bf bf = load_frag_B_rm_f32(Yb, Mm, k, (tj + t) * 16);
      acc[t] = wmma_bf16(a, bf, acc[t]);
    }
  }
#pragma unroll
  for (int t = 0; t < 4; ++t)
    store_tile_C(Qb, Mm, ti * 16, (tj + t) * 16, acc[t]);
}

// ---------------------------------------------------------------------------
// Stage 2: ImG = I - Q@Y1^T ; C2 = Q@Y2^T   (wave: 16x32 per output, 4 accs,
// shared A fragment across both matrices and both N-tiles)
// ---------------------------------------------------------------------------
__global__ void k_gemm_GC(const float* __restrict__ Q,
                          const float* __restrict__ Y1,
                          const float* __restrict__ Y2,
                          float* __restrict__ ImG,
                          float* __restrict__ C2) {
  const int TI = Dd / 16, TJ2 = Dd / 32;
  const int gw = (blockIdx.x * blockDim.x + threadIdx.x) >> 5;
  const int b  = gw / (TI * TJ2);
  const int r  = gw % (TI * TJ2);
  const int ti = r / TJ2, tj = (r % TJ2) * 2;   // tile-col base, 16-col units
  const float* Qb  = Q  + (size_t)b * Dd * Mm;
  const float* Y1b = Y1 + (size_t)b * Dd * Mm;
  const float* Y2b = Y2 + (size_t)b * Dd * Mm;
  float* Gb = ImG + (size_t)b * Dd * Dd;
  float* Cb = C2  + (size_t)b * Dd * Dd;

  v8f g[2] = {{}, {}};
  v8f c[2] = {{}, {}};
#pragma unroll 4
  for (int k = 0; k < Mm; k += 32) {
    v16bf a = load_frag_A_f32(Qb, Mm, ti * 16, k);
#pragma unroll
    for (int t = 0; t < 2; ++t) {
      v16bf b1 = load_frag_B_tr_f32(Y1b, Mm, k, (tj + t) * 16);
      g[t] = wmma_bf16(a, b1, g[t]);
      v16bf b2 = load_frag_B_tr_f32(Y2b, Mm, k, (tj + t) * 16);
      c[t] = wmma_bf16(a, b2, c[t]);
    }
  }
  const int lane = threadIdx.x & 31;
  const int rb   = ti * 16 + ((lane >> 4) << 3);
#pragma unroll
  for (int t = 0; t < 2; ++t) {
    const int col = (tj + t) * 16 + (lane & 15);
#pragma unroll
    for (int v = 0; v < 8; ++v) {
      const int row = rb + v;
      const float idv = (row == col) ? 1.0f : 0.0f;
      Gb[(size_t)row * Dd + col] = idv - g[t][v];
      Cb[(size_t)row * Dd + col] = c[t][v];
    }
  }
}

// ---------------------------------------------------------------------------
// Stage 3: per-batch vectors: Piy1 = P@y1, y1tP = y1^T@P, denom, u = y1/denom
// ---------------------------------------------------------------------------
__global__ void k_vecs(const float* __restrict__ P,
                       const float* __restrict__ y1,
                       float* __restrict__ Piy1, float* __restrict__ y1tP,
                       float* __restrict__ u, float* __restrict__ denom) {
  const int b = blockIdx.x, j = threadIdx.x;
  const float* Pb = P + (size_t)b * Dd * Dd;
  const float* y  = y1 + (size_t)b * Dd;
  float s1 = 0.f, s2 = 0.f;
  for (int i = 0; i < Dd; ++i) {
    s1 += Pb[(size_t)j * Dd + i] * y[i];  // (P y1)[j]
    s2 += y[i] * Pb[(size_t)i * Dd + j];  // (y1^T P)[j]
  }
  Piy1[b * Dd + j] = s1;
  y1tP[b * Dd + j] = s2;

  __shared__ float red[Dd];
  red[j] = y[j] * s1;
  __syncthreads();
  for (int st = Dd / 2; st > 0; st >>= 1) {
    if (j < st) red[j] += red[j + st];
    __syncthreads();
  }
  const float den = 1.0f + red[0];
  if (j == 0) denom[b] = den;
  u[b * Dd + j] = y[j] / den;
}

// ---------------------------------------------------------------------------
// Stage 4: P_inv_new = P - Piy1 (y1^T P) / denom   -> second output
// ---------------------------------------------------------------------------
__global__ void k_pinv(const float* __restrict__ P,
                       const float* __restrict__ Piy1,
                       const float* __restrict__ y1tP,
                       const float* __restrict__ denom,
                       float* __restrict__ Pout) {
  const int b = blockIdx.x, j = threadIdx.x;
  const float inv_den = 1.0f / denom[b];
  const float* Pb = P + (size_t)b * Dd * Dd;
  float* Ob = Pout + (size_t)b * Dd * Dd;
  for (int i = 0; i < Dd; ++i) {
    Ob[(size_t)i * Dd + j] =
        Pb[(size_t)i * Dd + j] - Piy1[b * Dd + i] * y1tP[b * Dd + j] * inv_den;
  }
}

// ---------------------------------------------------------------------------
// Stage 5: rank-1 fixes:
//   W_bf = bf16( ImG - u (y1^T ImG) )   (W is consumed only as a WMMA A-operand)
//   B    = C2 - u (y1^T C2) + u y2^T    (f32, in place; accumulator preload)
// ---------------------------------------------------------------------------
__global__ void k_wb(const float* __restrict__ ImG, float* __restrict__ C2,
                     const float* __restrict__ u,
                     const float* __restrict__ y1,
                     const float* __restrict__ y2,
                     bf16_t* __restrict__ Wbf) {
  const int b = blockIdx.x, j = threadIdx.x;
  const float* Gb = ImG + (size_t)b * Dd * Dd;
  float*       Cb = C2  + (size_t)b * Dd * Dd;
  bf16_t*      Wb = Wbf + (size_t)b * Dd * Dd;
  const float* y  = y1 + (size_t)b * Dd;
  const float* ub = u  + (size_t)b * Dd;
  float r1 = 0.f, r2 = 0.f;
  for (int i = 0; i < Dd; ++i) {
    r1 += y[i] * Gb[(size_t)i * Dd + j];
    r2 += y[i] * Cb[(size_t)i * Dd + j];
  }
  const float y2j = y2[(size_t)b * Dd + j];
  for (int i = 0; i < Dd; ++i) {
    const float ui = ub[i];
    Wb[(size_t)i * Dd + j] = (__bf16)(Gb[(size_t)i * Dd + j] - ui * r1);
    Cb[(size_t)i * Dd + j] += ui * (y2j - r2);
  }
}

// ---------------------------------------------------------------------------
// Flat f32 -> bf16 copy. The iterate is stored COLUMN-major, and A0 = A_start^T,
// so A0 column-major == A_start row-major: a plain flat convert, no transpose.
// ---------------------------------------------------------------------------
__global__ void k_cvt_bf(const float* __restrict__ src, bf16_t* __restrict__ dst,
                         int n) {
  const int i = blockIdx.x * blockDim.x + threadIdx.x;
  if (i < n) dst[i] = (__bf16)src[i];
}

// ---------------------------------------------------------------------------
// Stage 6 (x10): S_next = W @ S_cur + B.  Block = 4 waves -> 64x64 output.
// The 64-column B panel of S_cur (col-major => one contiguous 32 KB region)
// is DMA'd into LDS once per block via the Tensor Data Mover, then all four
// waves read B fragments from LDS (4x reuse).  A fragments (W rows) and the
// f32 accumulator preload (B matrix) come from global.  K-loop fully unrolled:
// 32 straight-line WMMAs per wave.  Final step stores f32 transposed to d_out.
// ---------------------------------------------------------------------------
template <bool FINAL>
__global__ void __launch_bounds__(128)
k_step(const bf16_t* __restrict__ W, const float* __restrict__ Bm,
       const bf16_t* __restrict__ Acur, bf16_t* __restrict__ Anext,
       float* __restrict__ outA) {
  __shared__ bf16_t ldsB[64 * Dd];          // 32 KB: cols 0..63, col-major

  const int wave = threadIdx.x >> 5;
  const int TI4 = Dd / 64, TJ4 = Dd / 64;   // 4 x 4 blocks of 64x64 per batch
  const int b  = blockIdx.x / (TI4 * TJ4);
  const int rr = blockIdx.x % (TI4 * TJ4);
  const int tib = (rr / TJ4) * 4;           // row-tile base (16-row units)
  const int tj  = (rr % TJ4) * 4;           // col-tile base (16-col units)

  const bf16_t* Wb = W    + (size_t)b * Dd * Dd;
  const float*  Cb = Bm   + (size_t)b * Dd * Dd;
  const bf16_t* Ab = Acur + (size_t)b * Dd * Dd;

  // ---- stage B panel (cols [tj*16, tj*16+64), all K) into LDS -------------
  const bf16_t* panel = Ab + (size_t)(tj * 16) * Dd;   // contiguous 32 KB
#if __has_builtin(__builtin_amdgcn_tensor_load_to_lds)
  if (wave == 0) {
    const unsigned long long ga = (unsigned long long)panel;
    const unsigned int lds_off  = (unsigned int)(unsigned long long)(&ldsB[0]);
    // D# group 0: count=1 | lds_addr | global_addr[56:0] | type=2
    v4u g0 = {1u, lds_off, (unsigned int)(ga & 0xFFFFFFFFu),
              ((unsigned int)((ga >> 32) & 0x01FFFFFFu)) | (2u << 30)};
    // D# group 1: data_size=2B; tensor_dim0=16384; tensor_dim1=1;
    //             tile_dim0=16384 (1D tile, tile_dim1=0); dim0_stride=16384
    v8i g1 = {(int)(1u << 16),        // [17:16] data_size = 1 (2 bytes)
              (int)(16384u << 16),    // tensor_dim0[15:0] << 16
              (int)(1u << 16),        // tensor_dim0 hi=0 | tensor_dim1[15:0]=1
              (int)(16384u << 16),    // tensor_dim1 hi=0 | tile_dim0=16384
              0,                      // tile_dim1=0 (unused), tile_dim2=0
              16384,                  // tensor_dim0_stride[31:0]
              0, 0};
    v4i gz = {0, 0, 0, 0};
#if __clang_major__ >= 23
    v8i g4 = {0, 0, 0, 0, 0, 0, 0, 0};
    __builtin_amdgcn_tensor_load_to_lds(g0, g1, gz, gz, g4, 0);
#else
    __builtin_amdgcn_tensor_load_to_lds(g0, g1, gz, gz, 0);
#endif
#if __has_builtin(__builtin_amdgcn_s_wait_tensorcnt)
    __builtin_amdgcn_s_wait_tensorcnt(0);
#endif
  }
  __syncthreads();
#else
  // Fallback: cooperative 128-bit copy (2048 x uint4 == 32 KB)
  {
    const uint4* s = (const uint4*)panel;
    uint4* d = (uint4*)ldsB;
    for (int i = threadIdx.x; i < 2048; i += 128) d[i] = s[i];
  }
  __syncthreads();
#endif

  // ---- compute: this wave owns rows [ti*16, ti*16+16), cols [tj*16, +64) --
  const int ti = tib + wave;
  v8f acc[4];
#pragma unroll
  for (int t = 0; t < 4; ++t)
    acc[t] = load_tile_C(Cb, Dd, ti * 16, (tj + t) * 16);

#pragma unroll
  for (int k = 0; k < Dd; k += 32) {
    v16bf a = load_frag_A_bf(Wb, Dd, ti * 16, k);
#pragma unroll
    for (int t = 0; t < 4; ++t) {
      v16bf bb = load_frag_B_tr_bf(&ldsB[0], Dd, k, t * 16);  // LDS-local cols
      acc[t] = wmma_bf16(a, bb, acc[t]);
    }
  }

  if (FINAL) {
    float* Ob = outA + (size_t)b * Dd * Dd;
#pragma unroll
    for (int t = 0; t < 4; ++t)
      store_tile_tr_f32(Ob, Dd, ti * 16, (tj + t) * 16, acc[t]);
  } else {
    bf16_t* Nb = Anext + (size_t)b * Dd * Dd;
#pragma unroll
    for (int t = 0; t < 4; ++t)
      store_tile_tr_bf(Nb, Dd, ti * 16, (tj + t) * 16, acc[t]);
  }
}

// ---------------------------------------------------------------------------
// Host launcher
// ---------------------------------------------------------------------------
extern "C" void kernel_launch(void* const* d_in, const int* in_sizes, int n_in,
                              void* d_out, int out_size, void* d_ws, size_t ws_size,
                              hipStream_t stream) {
  (void)in_sizes; (void)n_in; (void)out_size; (void)ws_size;
  const float* A_start = (const float*)d_in[0];
  const float* Y1      = (const float*)d_in[1];
  const float* Y2      = (const float*)d_in[2];
  const float* y1      = (const float*)d_in[3];
  const float* y2      = (const float*)d_in[4];
  const float* P       = (const float*)d_in[5];
  // d_in[6] = h (unused by the math), d_in[7] = m_prox (== MPROX, hardcoded)

  float* out_A = (float*)d_out;                                   // BS*D*D
  float* out_P = (float*)d_out + (size_t)BSn * Dd * Dd;           // BS*D*D

  const size_t DDall = (size_t)BSn * Dd * Dd;
  float* ws    = (float*)d_ws;
  float* w_ImG = ws;                 // 1 unit f32
  float* w_C2  = ws + DDall;         // 1 unit f32 -> becomes B in place
  float* w_Q   = ws + 2 * DDall;     // 2 units f32 (BS*D*M); recycled below
  // After stage 2, Q is dead: recycle its 2 f32 units as 3 bf16 half-units.
  bf16_t* w_Wbf = (bf16_t*)w_Q;            // DDall bf16
  bf16_t* w_A0  = (bf16_t*)w_Q + DDall;    // DDall bf16
  bf16_t* w_A1  = (bf16_t*)w_Q + 2*DDall;  // DDall bf16 (fits: 3*0.5 < 2 units)
  float* w_vec  = ws + 4 * DDall;
  float* v_Piy1 = w_vec;
  float* v_y1tP = w_vec + (size_t)BSn * Dd;
  float* v_u    = w_vec + 2 * (size_t)BSn * Dd;
  float* v_den  = w_vec + 3 * (size_t)BSn * Dd;

  const int WAVES_PER_BLK = 4, BLK = 32 * WAVES_PER_BLK;

  // 1) Q = P @ Y1                (waves: 64 * 16 * 8)
  {
    int waves = BSn * (Dd / 16) * (Mm / 64);
    k_gemm_Q<<<dim3(waves / WAVES_PER_BLK), dim3(BLK), 0, stream>>>(P, Y1, w_Q);
  }
  // 2) ImG = I - Q@Y1^T ; C2 = Q@Y2^T   (waves: 64 * 16 * 8)
  {
    int waves = BSn * (Dd / 16) * (Dd / 32);
    k_gemm_GC<<<dim3(waves / WAVES_PER_BLK), dim3(BLK), 0, stream>>>(
        w_Q, Y1, Y2, w_ImG, w_C2);
  }
  // 3) vectors (Piy1, y1tP, denom, u)
  k_vecs<<<dim3(BSn), dim3(Dd), 0, stream>>>(P, y1, v_Piy1, v_y1tP, v_u, v_den);
  // 4) P_inv_new output
  k_pinv<<<dim3(BSn), dim3(Dd), 0, stream>>>(P, v_Piy1, v_y1tP, v_den, out_P);
  // 5) rank-1 fixes -> W_bf (bf16, in recycled Q region), B (f32, in C2)
  k_wb<<<dim3(BSn), dim3(Dd), 0, stream>>>(w_ImG, w_C2, v_u, y1, y2, w_Wbf);
  // 6) A0 (column-major bf16) = flat bf16 copy of A_start
  {
    int n = (int)DDall;
    k_cvt_bf<<<dim3((n + 255) / 256), dim3(256), 0, stream>>>(A_start, w_A0, n);
  }
  // 7) 10 x (S <- W S + B); TDM-staged B panel; last step stores f32 to d_out
  {
    dim3 grid(BSn * (Dd / 64) * (Dd / 64)), blk(128);   // 1024 blocks, 4 waves
    bf16_t* src = w_A0;
    bf16_t* dst = w_A1;
    for (int s = 0; s < MPROX - 1; ++s) {
      k_step<false><<<grid, blk, 0, stream>>>(w_Wbf, w_C2, src, dst, nullptr);
      bf16_t* t = src; src = dst; dst = t;
    }
    // final step: store f32, transposed back to the reference layout
    k_step<true><<<grid, blk, 0, stream>>>(w_Wbf, w_C2, src, nullptr, out_A);
  }
}
